// LSTM_attention_8985071583525
// MI455X (gfx1250) — compile-verified, matrix operand order checked by reading, not run on hardware
//
#include <hip/hip_runtime.h>
#include <hip/hip_bf16.h>
#include <math.h>

// ---------------------------------------------------------------------------
// Problem constants (reference): B=8, L=1024, D_MODEL=1024, HIDDEN=512,
// N_HEADS=8, dh=128.  M = B*L = 8192 rows.
// ---------------------------------------------------------------------------
#define BATCH   8
#define SEQ     1024
#define DMODEL  1024
#define HID     512
#define G4      2048           // 4*HID
#define NHEADS  8
#define DH      128
#define MROWS   (BATCH * SEQ)  // 8192

typedef __attribute__((ext_vector_type(16))) __bf16 v16bf;
typedef __attribute__((ext_vector_type(8)))  __bf16 v8bf;
typedef __attribute__((ext_vector_type(8)))  float  v8f;
typedef __attribute__((ext_vector_type(4)))  float  f4;

// ---------------------------------------------------------------------------
// CDNA5 WMMA helpers (v_wmma_f32_16x16x32_bf16).
// A 16x32: lane%16 = row, K-pattern for half h = lane/16: {8h+0..7, 16+8h+0..7}
// B 32x16: lane%16 = col n, same K ordering -> for C = A*Bw^T (Bw (N,K) row-
// major) the B fragment is "row n of Bw with the A K-pattern".
// C/D: lane holds column lane%16; element v is row 8*(lane/16)+v.
// ---------------------------------------------------------------------------
__device__ __forceinline__ v8f wmma_bf16(v16bf a, v16bf b, v8f c) {
  return __builtin_amdgcn_wmma_f32_16x16x32_bf16(false, a, false, b,
                                                 (short)0, c, false, false);
}

// bf16 fragment: two contiguous 16-byte chunks (b128 loads).
__device__ __forceinline__ v16bf frag_ld_bf16(const __bf16* __restrict__ p, int half) {
  const __bf16* q = p + 8 * half;
  v8bf lo = *(const v8bf*)(q);
  v8bf hi = *(const v8bf*)(q + 16);
  v16bf f;
#pragma unroll
  for (int j = 0; j < 8; ++j) { f[j] = lo[j]; f[8 + j] = hi[j]; }
  return f;
}

// ---------------------------------------------------------------------------
// fp32 -> bf16 bulk conversion (memory-bound, one pass). n % 8 == 0.
// ---------------------------------------------------------------------------
__global__ __launch_bounds__(256) void cvt_bf16_kernel(const float* __restrict__ in,
                                                       __bf16* __restrict__ out) {
  const size_t i = ((size_t)blockIdx.x * blockDim.x + threadIdx.x) * 8;
  f4 a = *(const f4*)(in + i);
  f4 b = *(const f4*)(in + i + 4);
  v8bf o;
#pragma unroll
  for (int j = 0; j < 4; ++j) { o[j] = (__bf16)a[j]; o[4 + j] = (__bf16)b[j]; }
  *(v8bf*)(out + i) = o;
}

// ---------------------------------------------------------------------------
// WMMA GEMM:  C(M,N) = act( A(M,K)bf16 * Bw(N,K)bf16^T + bias(N) )
// Block = 256 threads = 8 waves, wave grid 2(M) x 4(N), wave tile 16x64,
// block tile 32 x 256.  TRANSV: write C transposed as bf16 [BATCH][N][SEQ]
// (for the attention V operand; per-lane 8 contiguous bf16 -> b128 store).
// ---------------------------------------------------------------------------
template <bool RELU, bool BF16OUT, bool TRANSV>
__global__ __launch_bounds__(256) void gemm_bias_kernel(
    const __bf16* __restrict__ A, const __bf16* __restrict__ Bw,
    const float* __restrict__ bias, void* __restrict__ Cout,
    int M, int N, int K) {
  const int lane = threadIdx.x & 31;
  const int wave = threadIdx.x >> 5;
  const int half = lane >> 4;
  const int ln16 = lane & 15;
  const int rowBase = blockIdx.y * 32 + (wave & 1) * 16;
  const int colBase = blockIdx.x * 256 + (wave >> 1) * 64;

  v8f acc[4] = {};
  const __bf16* Arow = A + (size_t)(rowBase + ln16) * K;

  for (int k0 = 0; k0 < K; k0 += 32) {
    __builtin_prefetch(Arow + k0 + 256, 0, 0);       // global_prefetch_b8
    v16bf afrag = frag_ld_bf16(Arow + k0, half);
#pragma unroll
    for (int t = 0; t < 4; ++t) {
      const __bf16* Brow = Bw + (size_t)(colBase + t * 16 + ln16) * K + k0;
      acc[t] = wmma_bf16(afrag, frag_ld_bf16(Brow, half), acc[t]);
    }
  }

#pragma unroll
  for (int t = 0; t < 4; ++t) {
    const int n = colBase + t * 16 + ln16;
    const float bv = bias[n];
#pragma unroll
    for (int v = 0; v < 8; ++v) {
      const int m = rowBase + 8 * half + v;
      float val = acc[t][v] + bv;
      if (RELU) val = fmaxf(val, 0.0f);
      if (TRANSV) {
        const int bb = m >> 10;              // m = b*SEQ + q, SEQ = 1024
        const int q  = m & (SEQ - 1);        // q contiguous in v -> b128 store
        ((__bf16*)Cout)[((size_t)bb * N + n) * SEQ + q] = (__bf16)val;
      } else if (BF16OUT) {
        ((__bf16*)Cout)[(size_t)m * N + n] = (__bf16)val;
      } else {
        ((float*)Cout)[(size_t)m * N + n] = val;
      }
    }
  }
}

// ---------------------------------------------------------------------------
// Sequential LSTM scan (fp32, latency-critical path).  grid = (BATCH, 2),
// 512 threads: thread j owns hidden unit j (all 4 gates).  h lives in LDS,
// Whh streams from the 192 MB L2 (8 MB resident).
// ---------------------------------------------------------------------------
__global__ __launch_bounds__(512) void lstm_scan_kernel(
    const float* __restrict__ xW_f, const float* __restrict__ xW_b,
    const float* __restrict__ Whh_f, const float* __restrict__ Whh_b,
    float* __restrict__ lstm) {
  const int b   = blockIdx.x;
  const int dir = blockIdx.y;
  const float* xW  = dir ? xW_b  : xW_f;
  const float* Whh = dir ? Whh_b : Whh_f;
  const int j = threadIdx.x;

  __shared__ float hsh[HID];
  hsh[j] = 0.0f;
  float c = 0.0f;

  const float* Wi = Whh + (size_t)(j)         * HID;
  const float* Wf = Whh + (size_t)(HID + j)   * HID;
  const float* Wg = Whh + (size_t)(2*HID + j) * HID;
  const float* Wo = Whh + (size_t)(3*HID + j) * HID;
  __syncthreads();

  for (int t = 0; t < SEQ; ++t) {
    const int tt = dir ? (SEQ - 1 - t) : t;
    const float* g4 = xW + ((size_t)b * SEQ + tt) * G4;
    float ai = g4[j], af = g4[HID + j], ag = g4[2*HID + j], ao = g4[3*HID + j];
#pragma unroll 4
    for (int k = 0; k < HID; ++k) {
      const float hk = hsh[k];
      ai = fmaf(Wi[k], hk, ai);
      af = fmaf(Wf[k], hk, af);
      ag = fmaf(Wg[k], hk, ag);
      ao = fmaf(Wo[k], hk, ao);
    }
    const float ig = 1.0f / (1.0f + __expf(-ai));
    const float fg = 1.0f / (1.0f + __expf(-af));
    const float gg = tanhf(ag);
    const float og = 1.0f / (1.0f + __expf(-ao));
    c = fg * c + ig * gg;
    const float hnew = og * tanhf(c);

    __syncthreads();
    hsh[j] = hnew;
    lstm[((size_t)b * SEQ + tt) * DMODEL + dir * HID + j] = hnew;
    __syncthreads();
  }
}

// ---------------------------------------------------------------------------
// qmask[b,q] = sign(|sum_d lstm[b,q,d]|)  (0 or 1)
// ---------------------------------------------------------------------------
__global__ __launch_bounds__(256) void qmask_kernel(const float* __restrict__ lstm,
                                                    float* __restrict__ qm) {
  const int r = blockIdx.x * blockDim.x + threadIdx.x;
  const float* p = lstm + (size_t)r * DMODEL;
  float s = 0.0f;
  for (int i = 0; i < DMODEL; i += 4) s += p[i] + p[i + 1] + p[i + 2] + p[i + 3];
  qm[r] = (s != 0.0f) ? 1.0f : 0.0f;
}

// ---------------------------------------------------------------------------
// d_out[:, :, 0:1024] = lstm_out (float4 copy into concat layout)
// ---------------------------------------------------------------------------
__global__ __launch_bounds__(256) void copy_lstm_kernel(const float* __restrict__ lstm,
                                                        float* __restrict__ out) {
  const size_t idx = (size_t)blockIdx.x * blockDim.x + threadIdx.x;
  const size_t r = idx >> 8;
  const int c4 = (int)(idx & 255);
  f4 v = *(const f4*)(lstm + r * DMODEL + c4 * 4);
  *(f4*)(out + r * (2 * DMODEL) + c4 * 4) = v;
}

// ---------------------------------------------------------------------------
// Flash attention: one wave per 16-row Q tile per (head, batch).
// Block = 128 threads (4 waves), grid = (SEQ/64, NHEADS, BATCH).
// 16 WMMAs per 32-key step (8 scores + 8 P@V).  V is pre-transposed
// [b][dh][key] so its fragments are contiguous b128 loads along keys.
// Score->A relayout goes through per-wave LDS guarded by s_wait_dscnt.
// Writes d_out[:, :, 1024:2048] = head_out * qmask + queries.
// ---------------------------------------------------------------------------
__global__ __launch_bounds__(128) void attention_kernel(
    const __bf16* __restrict__ Qb, const __bf16* __restrict__ Kb,
    const __bf16* __restrict__ Vt, const float* __restrict__ lstm,
    const float* __restrict__ qmask, float* __restrict__ out) {
  const int lane = threadIdx.x & 31;
  const int wave = threadIdx.x >> 5;
  const int half = lane >> 4;
  const int ln16 = lane & 15;
  const int b = blockIdx.z;
  const int h = blockIdx.y;
  const int q0 = (blockIdx.x * 4 + wave) * 16;

  __shared__ float pbuf[4][16 * 32];
  float* pw = pbuf[wave];

  const float scale = 0.0883883476483184f;  // 1/sqrt(128)

  v16bf qf[4];
  const __bf16* qrow = Qb + ((size_t)(b * SEQ + q0 + ln16)) * DMODEL + h * DH;
#pragma unroll
  for (int kk = 0; kk < 4; ++kk) qf[kk] = frag_ld_bf16(qrow + kk * 32, half);

  v8f O[8] = {};
  float mrun[8], lrun[8];
#pragma unroll
  for (int v = 0; v < 8; ++v) { mrun[v] = -1e30f; lrun[v] = 0.0f; }

  for (int kt = 0; kt < SEQ; kt += 32) {
    // scores: S0 = Q @ K[kt..kt+15]^T, S1 = Q @ K[kt+16..kt+31]^T
    v8f S0 = {}, S1 = {};
#pragma unroll
    for (int kk = 0; kk < 4; ++kk) {
      const __bf16* kr0 = Kb + ((size_t)(b * SEQ + kt + ln16)) * DMODEL + h * DH + kk * 32;
      S0 = wmma_bf16(qf[kk], frag_ld_bf16(kr0, half), S0);
      S1 = wmma_bf16(qf[kk], frag_ld_bf16(kr0 + (size_t)16 * DMODEL, half), S1);
    }

    // online softmax (element v = row v + 8*half; rows span a 16-lane group)
    float p0[8], p1[8];
#pragma unroll
    for (int v = 0; v < 8; ++v) {
      const float s0 = S0[v] * scale;
      const float s1 = S1[v] * scale;
      float rmax = fmaxf(s0, s1);
      for (int msk = 1; msk < 16; msk <<= 1)
        rmax = fmaxf(rmax, __shfl_xor(rmax, msk, 32));
      const float mnew = fmaxf(mrun[v], rmax);
      const float e0 = __expf(s0 - mnew);
      const float e1 = __expf(s1 - mnew);
      float rsum = e0 + e1;
      for (int msk = 1; msk < 16; msk <<= 1)
        rsum += __shfl_xor(rsum, msk, 32);
      const float corr = __expf(mrun[v] - mnew);
      lrun[v] = lrun[v] * corr + rsum;
      mrun[v] = mnew;
      p0[v] = e0; p1[v] = e1;
#pragma unroll
      for (int nn = 0; nn < 8; ++nn) O[nn][v] *= corr;
    }

    // P relayout: C-fragment -> A-fragment through wave-private LDS
#pragma unroll
    for (int v = 0; v < 8; ++v) {
      const int r = v + 8 * half;
      pw[r * 32 + ln16]      = p0[v];
      pw[r * 32 + 16 + ln16] = p1[v];
    }
    asm volatile("s_wait_dscnt 0x0" ::: "memory");  // DS RAW within wave
    v16bf pf;
    {
      const float* pr = pw + ln16 * 32;
#pragma unroll
      for (int j = 0; j < 8; ++j) {
        pf[j]     = (__bf16)pr[8 * half + j];
        pf[8 + j] = (__bf16)pr[16 + 8 * half + j];
      }
    }

    // O += P @ V[kt..kt+31, :]  — Vt rows are contiguous along keys
#pragma unroll
    for (int nn = 0; nn < 8; ++nn) {
      const __bf16* vp = Vt + ((size_t)b * DMODEL + h * DH + nn * 16 + ln16) * SEQ + kt;
      O[nn] = wmma_bf16(pf, frag_ld_bf16(vp, half), O[nn]);
    }
  }

  // epilogue: normalize, mask, residual, write concat output
#pragma unroll
  for (int v = 0; v < 8; ++v) {
    const int row = q0 + 8 * half + v;
    const float inv_l = qmask[b * SEQ + row] / lrun[v];
#pragma unroll
    for (int nn = 0; nn < 8; ++nn) {
      const int col = h * DH + nn * 16 + ln16;
      const float val = O[nn][v] * inv_l +
                        lstm[((size_t)(b * SEQ + row)) * DMODEL + col];
      out[((size_t)(b * SEQ + row)) * (2 * DMODEL) + DMODEL + col] = val;
    }
  }
}

// ---------------------------------------------------------------------------
// Workspace layout (bytes):
//   [  0MB) xW_f   8192*2048*4 = 64 MB   } after the scan this region is dead
//   [ 64MB) xW_b   8192*2048*4 = 64 MB   } and is reused for Qb/Kb/Vt (48 MB)
//   [128MB) lstm   8192*1024*4 = 32 MB
//   [160MB) qmask  8192*4 (padded to 1 MB)
//   [161MB) xb     8192*1024*2 = 16 MB   (x in bf16)
//   [177MB) labb   16 MB                 (label_embs in bf16)
//   [193MB) lstmb  16 MB                 (lstm_out in bf16)
//   [209MB) Wf2 4MB | Wb2 4MB | Wq2 2MB | Wk2 2MB | Wv2 2MB   (bf16 weights)
//   total ~223 MB
// ---------------------------------------------------------------------------
extern "C" void kernel_launch(void* const* d_in, const int* in_sizes, int n_in,
                              void* d_out, int out_size, void* d_ws, size_t ws_size,
                              hipStream_t stream) {
  (void)in_sizes; (void)n_in; (void)out_size; (void)ws_size;
  const float* x     = (const float*)d_in[0];
  const float* lab   = (const float*)d_in[1];
  const float* Wih_f = (const float*)d_in[2];
  const float* Whh_f = (const float*)d_in[3];
  const float* b_f   = (const float*)d_in[4];
  const float* Wih_b = (const float*)d_in[5];
  const float* Whh_b = (const float*)d_in[6];
  const float* b_b   = (const float*)d_in[7];
  const float* Wq    = (const float*)d_in[8];
  const float* bq    = (const float*)d_in[9];
  const float* Wk    = (const float*)d_in[10];
  const float* bk    = (const float*)d_in[11];
  const float* Wv    = (const float*)d_in[12];
  const float* bv    = (const float*)d_in[13];
  float* out = (float*)d_out;

  char* wsb = (char*)d_ws;
  const size_t MB = 1u << 20;
  float*  xWf   = (float*)(wsb);
  float*  xWb   = (float*)(wsb + 64 * MB);
  float*  lstm  = (float*)(wsb + 128 * MB);
  float*  qm    = (float*)(wsb + 160 * MB);
  __bf16* xb    = (__bf16*)(wsb + 161 * MB);
  __bf16* labb  = (__bf16*)(wsb + 177 * MB);
  __bf16* lstmb = (__bf16*)(wsb + 193 * MB);
  __bf16* Wf2   = (__bf16*)(wsb + 209 * MB);
  __bf16* Wb2   = (__bf16*)(wsb + 213 * MB);
  __bf16* Wq2   = (__bf16*)(wsb + 217 * MB);
  __bf16* Wk2   = (__bf16*)(wsb + 219 * MB);
  __bf16* Wv2   = (__bf16*)(wsb + 221 * MB);
  // aliases into the xW region (dead after the scan):
  __bf16* Qb = (__bf16*)xWf;
  __bf16* Kb = Qb + (size_t)MROWS * DMODEL;
  __bf16* Vt = Kb + (size_t)MROWS * DMODEL;

  const dim3 tb(256);
  const int CVT = 256 * 8;  // elements per cvt block

  // 0) one-shot bf16 conversions (activations + weights)
  cvt_bf16_kernel<<<(MROWS * DMODEL) / CVT, 256, 0, stream>>>(x, xb);
  cvt_bf16_kernel<<<(MROWS * DMODEL) / CVT, 256, 0, stream>>>(lab, labb);
  cvt_bf16_kernel<<<(G4 * DMODEL) / CVT, 256, 0, stream>>>(Wih_f, Wf2);
  cvt_bf16_kernel<<<(G4 * DMODEL) / CVT, 256, 0, stream>>>(Wih_b, Wb2);
  cvt_bf16_kernel<<<(DMODEL * DMODEL) / CVT, 256, 0, stream>>>(Wq, Wq2);
  cvt_bf16_kernel<<<(DMODEL * DMODEL) / CVT, 256, 0, stream>>>(Wk, Wk2);
  cvt_bf16_kernel<<<(DMODEL * DMODEL) / CVT, 256, 0, stream>>>(Wv, Wv2);

  // 1) input projections (WMMA bf16, f32 out)
  gemm_bias_kernel<false, false, false><<<dim3(G4 / 256, MROWS / 32), tb, 0, stream>>>(
      xb, Wf2, b_f, xWf, MROWS, G4, DMODEL);
  gemm_bias_kernel<false, false, false><<<dim3(G4 / 256, MROWS / 32), tb, 0, stream>>>(
      xb, Wb2, b_b, xWb, MROWS, G4, DMODEL);

  // 2) sequential bidirectional scan (fp32 VALU)
  lstm_scan_kernel<<<dim3(BATCH, 2), 512, 0, stream>>>(xWf, xWb, Whh_f, Whh_b, lstm);
  cvt_bf16_kernel<<<(MROWS * DMODEL) / CVT, 256, 0, stream>>>(lstm, lstmb);

  // 3) Q/K/V projections with ReLU (V written transposed for attention)
  gemm_bias_kernel<true, true, false><<<dim3(DMODEL / 256, MROWS / 32), tb, 0, stream>>>(
      lstmb, Wq2, bq, Qb, MROWS, DMODEL, DMODEL);
  gemm_bias_kernel<true, true, false><<<dim3(DMODEL / 256, MROWS / 32), tb, 0, stream>>>(
      labb, Wk2, bk, Kb, MROWS, DMODEL, DMODEL);
  gemm_bias_kernel<true, true, true><<<dim3(DMODEL / 256, MROWS / 32), tb, 0, stream>>>(
      labb, Wv2, bv, Vt, MROWS, DMODEL, DMODEL);

  // 4) qmask + lstm_out into concat output
  qmask_kernel<<<MROWS / 256, 256, 0, stream>>>(lstm, qm);
  copy_lstm_kernel<<<(MROWS * (DMODEL / 4)) / 256, 256, 0, stream>>>(lstm, out);

  // 5) flash attention -> d_out[:, :, 1024:2048]
  attention_kernel<<<dim3(SEQ / 64, NHEADS, BATCH), 128, 0, stream>>>(
      Qb, Kb, Vt, lstm, qm, out);
}
